// GRU_34548716929347
// MI455X (gfx1250) — compile-verified, matrix operand order checked by reading, not run on hardware
//
#include <hip/hip_runtime.h>
#include <cstdint>

#define T_SZ 32768
#define INSZ 256
#define HID  512
#define G3H  1536

#define NWG      32   // persistent workgroups in the scan
#define UPW      16   // hidden units per WG
#define ROWS     48   // 3 * UPW rows of W_hh per WG
#define STHREADS 384  // 8 lanes per row * 48 rows

typedef _Float16 half_t;
typedef half_t v16h __attribute__((ext_vector_type(16)));
typedef half_t v8h  __attribute__((ext_vector_type(8)));
typedef half_t v4h  __attribute__((ext_vector_type(4)));
typedef float  v8f  __attribute__((ext_vector_type(8)));

// ---------------------------------------------------------------------------
// f32 -> f16 conversion (bandwidth-bound, vectorized)
// ---------------------------------------------------------------------------
__global__ __launch_bounds__(256) void convert_f16(const float* __restrict__ src,
                                                   half_t* __restrict__ dst, int n4) {
    int i = blockIdx.x * blockDim.x + threadIdx.x;
    const int stride = gridDim.x * blockDim.x;
    for (; i < n4; i += stride) {
        const float4 v = ((const float4*)src)[i];
        v4h h = { (half_t)v.x, (half_t)v.y, (half_t)v.z, (half_t)v.w };
        ((v4h*)dst)[i] = h;
    }
}

// ---------------------------------------------------------------------------
// Phase 1: igates = input @ W_ih^T + bias  (f16 WMMA, f32 accumulate)
// Block: 256 threads = 8 waves; tile 128(M) x 64(N); full K=256 panel of B
// staged once in LDS via async global->LDS; A fragments stream from global.
// Inner loop: 4 WMMAs per 32-K step, no barriers.
// ---------------------------------------------------------------------------
__global__ __launch_bounds__(256) void igates_gemm(
    const half_t* __restrict__ Ah,   // input  f16: T x 256
    const half_t* __restrict__ Wh,   // W_ih   f16: 1536 x 256
    const float*  __restrict__ bias, // 1536
    float* __restrict__ C)           // igates: T x 1536
{
    __shared__ half_t Bt[64][INSZ];  // Bt[n][k] = Wh[n0+n][k], 32 KB

    const int tid   = threadIdx.x;
    const int n0    = blockIdx.x * 64;
    const int m0    = blockIdx.y * 128;
    const int lane  = tid & 31;
    const int wid   = tid >> 5;      // 0..7 : 16-row M sub-tile per wave
    const int lrow  = lane & 15;     // A row / B,C column within 16x16 tile
    const int khalf = lane >> 4;     // K half-group select

    // Async-copy the 64x256 f16 B panel into LDS (ASYNCcnt-tracked).
    // LDS byte address taken from the real pointer: generic LDS pointers on
    // gfx1250 carry the LDS offset in addr[31:0] (flat aperture mapping), and
    // passing it into the asm captures Bt so its loads cannot be folded away.
    {
        const int row = tid >> 2;              // 0..63
        const int seg = (tid & 3) * 64;        // f16 element offset in row
        const unsigned ldsoff = (unsigned)(uintptr_t)&Bt[row][seg];
        const unsigned goff =
            (unsigned)(((n0 + row) * INSZ + seg) * (int)sizeof(half_t));
        const unsigned long long gbase = (unsigned long long)(uintptr_t)Wh;
#pragma unroll
        for (int i = 0; i < 8; ++i) {
            unsigned l = ldsoff + (unsigned)(i * 16);
            unsigned g = goff + (unsigned)(i * 16);
            asm volatile("global_load_async_to_lds_b128 %0, %1, %2 offset:0"
                         :: "v"(l), "v"(g), "s"(gbase) : "memory");
        }
    }
    asm volatile("s_wait_asynccnt 0x0" ::: "memory");
    __syncthreads();

    v8f acc[4] = {};
    const half_t* arow = Ah + (size_t)(m0 + wid * 16 + lrow) * INSZ;

#pragma unroll 4
    for (int k0 = 0; k0 < INSZ; k0 += 32) {
        // A fragment: two contiguous 16B runs per lane (global b128 loads)
        const v8h alo = *(const v8h*)(arow + k0 + khalf * 8);
        const v8h ahi = *(const v8h*)(arow + k0 + 16 + khalf * 8);
        const v16h av = __builtin_shufflevector(
            alo, ahi, 0, 1, 2, 3, 4, 5, 6, 7, 8, 9, 10, 11, 12, 13, 14, 15);
#pragma unroll
        for (int j = 0; j < 4; ++j) {
            const half_t* brow = &Bt[j * 16 + lrow][0];
            const v8h blo = *(const v8h*)(brow + k0 + khalf * 8);
            const v8h bhi = *(const v8h*)(brow + k0 + 16 + khalf * 8);
            const v16h bv = __builtin_shufflevector(
                blo, bhi, 0, 1, 2, 3, 4, 5, 6, 7, 8, 9, 10, 11, 12, 13, 14, 15);
            acc[j] = __builtin_amdgcn_wmma_f32_16x16x32_f16(
                         false, av, false, bv, (short)0, acc[j], false, false);
        }
    }

    // C layout: VGPR r -> M = r + 8*khalf, N = lane&15
#pragma unroll
    for (int j = 0; j < 4; ++j) {
        const int n = n0 + j * 16 + lrow;
        const float b = bias[n];
#pragma unroll
        for (int r = 0; r < 8; ++r) {
            const int m = m0 + wid * 16 + r + 8 * khalf;
            C[(size_t)m * G3H + n] = acc[j][r] + b;
        }
    }
}

// ---------------------------------------------------------------------------
// Phase 2: persistent GRU scan. 32 WGs x 384 threads; each WG owns the r/z/n
// rows of W_hh for 16 hidden units, held f16 in LDS. h double-buffered in
// global (L2-resident); device-scope generation barrier between steps.
// ---------------------------------------------------------------------------
__global__ __launch_bounds__(STHREADS) void gru_scan(
    const float* __restrict__ input,
    const float* __restrict__ h0,
    const float* __restrict__ Wih,
    const float* __restrict__ Whh,
    const float* __restrict__ bias,
    const float* __restrict__ bias_n,
    const float* __restrict__ ig,      // precomputed igates or nullptr
    float* __restrict__ hbuf,          // 2*HID floats (double buffer)
    unsigned* __restrict__ barrier,    // [0]=count [1]=generation
    float* __restrict__ out,           // T floats
    int use_ig)
{
    __shared__ half_t Wsh[ROWS * HID];
    __shared__ float  hsh[HID];
    __shared__ float  hgs[ROWS];
    __shared__ float  igs[ROWS];
    __shared__ float  xsh[INSZ];

    const int tid = threadIdx.x;
    const int wg  = blockIdx.x;

    // Stage this WG's W_hh slice into LDS (f16), once.
    for (int i = tid; i < ROWS * HID; i += STHREADS) {
        const int lr = i / HID, k = i % HID;
        const int gr = (lr / UPW) * HID + wg * UPW + (lr % UPW);
        Wsh[i] = (half_t)Whh[(size_t)gr * HID + k];
    }
    __syncthreads();

    const int r   = tid >> 3;  // 0..47: local row
    const int sub = tid & 7;   // 8 lanes cooperate per row

    for (int t = 0; t < T_SZ; ++t) {
        const float* hp = (t == 0) ? h0 : (hbuf + ((t - 1) & 1) * HID);
        for (int i = tid; i < HID; i += STHREADS) hsh[i] = hp[i];
        if (!use_ig)
            for (int i = tid; i < INSZ; i += STHREADS)
                xsh[i] = input[(size_t)t * INSZ + i];
        __syncthreads();

        // hg = W_hh_slice @ h : 48 dots of length 512, 8 lanes each.
        // Vectorized: v8h weight chunks (ds_load_b128) + float4 h chunks.
        {
            float s = 0.f;
            const v8h*    wv = (const v8h*)(&Wsh[r * HID]) + sub * 8;
            const float4* hv = (const float4*)(hsh) + sub * 16;
#pragma unroll
            for (int c = 0; c < 8; ++c) {
                const v8h   w  = wv[c];
                const float4 ha = hv[2 * c];
                const float4 hb = hv[2 * c + 1];
                s = fmaf((float)w[0], ha.x, s);
                s = fmaf((float)w[1], ha.y, s);
                s = fmaf((float)w[2], ha.z, s);
                s = fmaf((float)w[3], ha.w, s);
                s = fmaf((float)w[4], hb.x, s);
                s = fmaf((float)w[5], hb.y, s);
                s = fmaf((float)w[6], hb.z, s);
                s = fmaf((float)w[7], hb.w, s);
            }
            s += __shfl_xor(s, 1);
            s += __shfl_xor(s, 2);
            s += __shfl_xor(s, 4);
            if (sub == 0) hgs[r] = s;
        }
        // Fallback: compute this step's igates slice on the fly
        if (!use_ig) {
            const int gr = (r / UPW) * HID + wg * UPW + (r % UPW);
            float s = 0.f;
            const float* wrow = Wih + (size_t)gr * INSZ + sub * 32;
            const float* xv   = &xsh[sub * 32];
#pragma unroll 8
            for (int k = 0; k < 32; ++k) s = fmaf(wrow[k], xv[k], s);
            s += __shfl_xor(s, 1);
            s += __shfl_xor(s, 2);
            s += __shfl_xor(s, 4);
            if (sub == 0) igs[r] = s + bias[gr];
        }
        __syncthreads();

        // Gate math + h update for this WG's 16 hidden units
        if (tid < UPW) {
            const int u = tid;
            const int j = wg * UPW + u;
            float ir, iz, inn;
            if (use_ig) {
                const float* igrow = ig + (size_t)t * G3H;
                ir  = igrow[j];
                iz  = igrow[HID + j];
                inn = igrow[2 * HID + j];
                __builtin_prefetch(igrow + G3H + j, 0, 1);  // next-step row
            } else {
                ir = igs[u]; iz = igs[UPW + u]; inn = igs[2 * UPW + u];
            }
            const float hr = hgs[u], hz = hgs[UPW + u], hn = hgs[2 * UPW + u];
            const float reset = 1.f / (1.f + __expf(-(ir + hr)));
            const float upd   = 1.f / (1.f + __expf(-(iz + hz)));
            const float nn    = tanhf(inn + reset * (hn + bias_n[j]));
            const float hold  = hsh[j];
            const float hnew  = nn + upd * (hold - nn);
            hbuf[(t & 1) * HID + j] = hnew;
            if (j == 0) out[t] = hnew;
            __threadfence();  // make h_new visible device-wide before barrier
        }
        __syncthreads();

        // Device-scope generation barrier across the 32 persistent WGs
        if (tid == 0) {
            const unsigned g0 = __hip_atomic_load(&barrier[1], __ATOMIC_RELAXED,
                                                  __HIP_MEMORY_SCOPE_AGENT);
            const unsigned prev = __hip_atomic_fetch_add(&barrier[0], 1u,
                                                         __ATOMIC_ACQ_REL,
                                                         __HIP_MEMORY_SCOPE_AGENT);
            if (prev == NWG - 1) {
                __hip_atomic_store(&barrier[0], 0u, __ATOMIC_RELAXED,
                                   __HIP_MEMORY_SCOPE_AGENT);
                __hip_atomic_fetch_add(&barrier[1], 1u, __ATOMIC_RELEASE,
                                       __HIP_MEMORY_SCOPE_AGENT);
            } else {
                while (__hip_atomic_load(&barrier[1], __ATOMIC_ACQUIRE,
                                         __HIP_MEMORY_SCOPE_AGENT) == g0)
                    __builtin_amdgcn_s_sleep(1);
            }
        }
        __syncthreads();
    }
}

// ---------------------------------------------------------------------------
extern "C" void kernel_launch(void* const* d_in, const int* in_sizes, int n_in,
                              void* d_out, int out_size, void* d_ws, size_t ws_size,
                              hipStream_t stream) {
    const float* input  = (const float*)d_in[0];  // T x 256
    const float* h0     = (const float*)d_in[1];  // 512
    const float* Wih    = (const float*)d_in[2];  // 1536 x 256
    const float* Whh    = (const float*)d_in[3];  // 1536 x 512
    const float* bias   = (const float*)d_in[4];  // 1536
    const float* bias_n = (const float*)d_in[5];  // 512
    float* out = (float*)d_out;                   // T x 1

    const size_t IGB = (size_t)T_SZ * G3H * sizeof(float);   // 201 MB
    const size_t AHB = (size_t)T_SZ * INSZ * sizeof(half_t); // 16 MB
    const size_t WHB = (size_t)G3H * INSZ * sizeof(half_t);  // 768 KB
    const size_t off_ig = 8192;
    const size_t off_ah = off_ig + IGB;
    const size_t off_wh = off_ah + AHB;
    const size_t need   = off_wh + WHB;

    unsigned* bar  = (unsigned*)d_ws;                  // [0..255]
    float*    hbuf = (float*)((char*)d_ws + 256);      // 4 KB double buffer
    float*    igb  = (float*)((char*)d_ws + off_ig);
    half_t*   Ah   = (half_t*)((char*)d_ws + off_ah);
    half_t*   Wh   = (half_t*)((char*)d_ws + off_wh);
    const int use_ig = (ws_size >= need) ? 1 : 0;

    hipMemsetAsync(d_ws, 0, 256, stream);  // barrier count/gen

    if (use_ig) {
        convert_f16<<<2048, 256, 0, stream>>>(input, Ah, T_SZ * INSZ / 4);
        convert_f16<<<384, 256, 0, stream>>>(Wih, Wh, G3H * INSZ / 4);
        dim3 grid(G3H / 64, T_SZ / 128);
        igates_gemm<<<grid, 256, 0, stream>>>(Ah, Wh, bias, igb);
    }
    gru_scan<<<NWG, STHREADS, 0, stream>>>(input, h0, Wih, Whh, bias, bias_n,
                                           use_ig ? igb : nullptr,
                                           hbuf, bar, out, use_ig);
}
